// TMM_79413945303453
// MI455X (gfx1250) — matrix operand order repaired and checked by reading, not compiled, not run on hardware
//
#include <hip/hip_runtime.h>

// ---------------------------------------------------------------------------
// Per-class 3-layer MLP heads, fused, bf16 WMMA (gfx1250).
//   out[b,c,:] = relu(relu(x[b,:]@W1[c]+b1[c])@W2[c]+b2[c])@W3[c]+b3[c]
// One workgroup per class. h1 lives in LDS as bf16 (A-ready for layer 2).
// Weight stream (the 1.28 GB roofline term) is register-pipelined:
//   round r:  ds_store(regs_r) | prefetch regs_{r+1} | WMMA from wt
// Two 16-col output tiles per round share each A load (halves L2 A-traffic).
// Layer 3 (D->2) folded into layer-2 epilogue + 16-lane shuffle reduction.
// ---------------------------------------------------------------------------

typedef __attribute__((ext_vector_type(16))) __bf16 v16bf;
typedef __attribute__((ext_vector_type(2)))  __bf16 bf16x2;
typedef __attribute__((ext_vector_type(8)))  float  v8f;
typedef __attribute__((ext_vector_type(4)))  float  fx4;

#define B_ 256
#define C_ 512
#define F_ 768
#define D_ 512

#define WAVES   16
#define THREADS (WAVES * 32)

// LDS pitches (bf16 elements). pitch*2 multiple of 16 B -> aligned b128 chunks.
#define H1P 520   // h1 row pitch   (1040 B/row, 16B aligned, conflict-light)
#define WTP 776   // wt row pitch   (1552 B/row; 388 dw % 64 = 4 -> spread banks)

#define H1_ELEMS (B_ * H1P)        // 133120 bf16
#define WT_ELEMS (32 * WTP)        //  24832 bf16 (two 16-col tiles per round)
#define LDS_BYTES ((H1_ELEMS + WT_ELEMS) * 2)   // 315,904 B < 320 KB WGP LDS

__device__ __forceinline__ v16bf load_ab(const __bf16* p) {
    // Two aligned 16-byte LDS chunks -> one v16bf operand block.
    uint4 t0 = *(const uint4*)(p);
    uint4 t1 = *(const uint4*)(p + 16);
    v16bf a;
    __builtin_memcpy(&a, &t0, 16);
    __builtin_memcpy((char*)&a + 16, &t1, 16);
    return a;
}

// Prefetch one staging round of weights into registers.
// Unit u covers rows {k,k+1} x cols {n..n+3} of the 32-column slice:
//   k = 2*(u>>3), n = 4*(u&7).  Consecutive threads cover 128 B row segments.
template<int UNITS>
__device__ __forceinline__ void stage_gload(const float* __restrict__ Wc, int n0,
                                            int tid, fx4* r0, fx4* r1) {
    #pragma unroll
    for (int i = 0; i < UNITS; ++i) {
        const int u = tid + i * THREADS;
        const int k = (u >> 3) * 2;
        const int n = (u & 7) * 4;
        const float* p = Wc + (size_t)k * D_ + n0 + n;
        r0[i] = __builtin_nontemporal_load((const fx4*)p);        // row k
        r1[i] = __builtin_nontemporal_load((const fx4*)(p + D_)); // row k+1
    }
}

// Convert + store the prefetched round into the transposed LDS tile wt[n][k].
// Packs (k,k+1) into one bf16x2 -> single ds_store_b32 per pair.
template<int UNITS>
__device__ __forceinline__ void stage_store(__bf16* wt, int tid,
                                            const fx4* r0, const fx4* r1) {
    #pragma unroll
    for (int i = 0; i < UNITS; ++i) {
        const int u = tid + i * THREADS;
        const int k = (u >> 3) * 2;
        const int n = (u & 7) * 4;
        #pragma unroll
        for (int j = 0; j < 4; ++j) {
            bf16x2 p;
            p.x = (__bf16)r0[i][j];
            p.y = (__bf16)r1[i][j];
            *(bf16x2*)(wt + (n + j) * WTP + k) = p;
        }
    }
}

__global__ __launch_bounds__(THREADS, 1)
void TMM_79413945303453_kernel(const float* __restrict__ x,
                               const float* __restrict__ W1,
                               const float* __restrict__ b1,
                               const float* __restrict__ W2,
                               const float* __restrict__ b2,
                               const float* __restrict__ W3,
                               const float* __restrict__ b3,
                               float* __restrict__ out)
{
    extern __shared__ __bf16 smem[];
    __bf16* h1 = smem;              // [B_][H1P]  bf16 activations
    __bf16* wt = smem + H1_ELEMS;   // [32][WTP]  staged weight tile (transposed)

    const int c    = blockIdx.x;
    const int tid  = threadIdx.x;
    const int lane = tid & 31;
    const int wave = tid >> 5;
    const int col  = lane & 15;     // WMMA N column / A row selector
    const int hi   = lane >> 4;     // lane-group -> K sub-block
    const int mbase = wave << 4;    // 16-row batch strip owned by this wave
    const int row   = mbase + col;  // this lane's A row (batch index)

    const float* xrow = x  + (size_t)row * F_;
    const float* W1c  = W1 + (size_t)c * F_ * D_;
    const float* W2c  = W2 + (size_t)c * D_ * D_;

    float o0[8], o1[8];
    #pragma unroll
    for (int v = 0; v < 8; ++v) { o0[v] = 0.f; o1[v] = 0.f; }

    // ------------- Layer 1: h1 = relu(x @ W1[c] + b1[c]) -> LDS bf16 -------
    {
        fx4 r0[6], r1[6];                       // 768*32/512 = 48 elems/thread
        stage_gload<6>(W1c, 0, tid, r0, r1);    // prefetch round 0
        for (int r = 0; r < D_ / 32; ++r) {
            const int n0 = r * 32;
            __syncthreads();                    // wt free (prev compute done)
            stage_store<6>(wt, tid, r0, r1);    // commit round r to LDS
            __syncthreads();                    // wt ready
            if (r + 1 < D_ / 32)                // prefetch round r+1; loads
                stage_gload<6>(W1c, n0 + 32, tid, r0, r1);  // overlap WMMA

            v8f acc0 = {}, acc1 = {};
            #pragma unroll 2
            for (int k0 = 0; k0 < F_; k0 += 32) {
                // A: x row f32 -> bf16 (elems 0..7 = kc.., 8..15 = kc+16..,
                // kc = k0 + 8*hi per ISA 16-bit A layout); shared by 2 tiles.
                const fx4 a0 = *(const fx4*)(xrow + k0 + hi * 8);
                const fx4 a1 = *(const fx4*)(xrow + k0 + hi * 8 + 4);
                const fx4 a2 = *(const fx4*)(xrow + k0 + 16 + hi * 8);
                const fx4 a3 = *(const fx4*)(xrow + k0 + 16 + hi * 8 + 4);
                v16bf A;
                #pragma unroll
                for (int j = 0; j < 4; ++j) {
                    A[j]      = (__bf16)a0[j];
                    A[4 + j]  = (__bf16)a1[j];
                    A[8 + j]  = (__bf16)a2[j];
                    A[12 + j] = (__bf16)a3[j];
                }
                const v16bf B0 = load_ab(wt + col * WTP        + k0 + hi * 8);
                const v16bf B1 = load_ab(wt + (col + 16) * WTP + k0 + hi * 8);
                acc0 = __builtin_amdgcn_wmma_f32_16x16x32_bf16(
                           false, A, false, B0, (short)0, acc0, false, false);
                acc1 = __builtin_amdgcn_wmma_f32_16x16x32_bf16(
                           false, A, false, B1, (short)0, acc1, false, false);
            }
            const float bias0 = b1[(size_t)c * D_ + n0 + col];
            const float bias1 = b1[(size_t)c * D_ + n0 + 16 + col];
            #pragma unroll
            for (int v = 0; v < 8; ++v) {
                const int m = mbase + v + hi * 8;    // C layout row
                float h0 = acc0[v] + bias0; h0 = h0 > 0.f ? h0 : 0.f;
                float h1v = acc1[v] + bias1; h1v = h1v > 0.f ? h1v : 0.f;
                h1[m * H1P + n0 + col]      = (__bf16)h0;
                h1[m * H1P + n0 + 16 + col] = (__bf16)h1v;
            }
        }
    }

    // ------ Layer 2 + 3 fused: o += relu(h1 @ W2[c] + b2[c]) @ W3[c] -------
    {
        fx4 r0[4], r1[4];                       // 512*32/512 = 32 elems/thread
        stage_gload<4>(W2c, 0, tid, r0, r1);
        const __bf16* arow = h1 + row * H1P;
        for (int r = 0; r < D_ / 32; ++r) {
            const int n0 = r * 32;
            __syncthreads();                    // first iter also fences h1
            stage_store<4>(wt, tid, r0, r1);
            __syncthreads();
            if (r + 1 < D_ / 32)
                stage_gload<4>(W2c, n0 + 32, tid, r0, r1);

            v8f acc0 = {}, acc1 = {};
            #pragma unroll 2
            for (int k0 = 0; k0 < D_; k0 += 32) {
                const v16bf A  = load_ab(arow + k0 + hi * 8);   // chunks +0,+16
                const v16bf B0 = load_ab(wt + col * WTP        + k0 + hi * 8);
                const v16bf B1 = load_ab(wt + (col + 16) * WTP + k0 + hi * 8);
                acc0 = __builtin_amdgcn_wmma_f32_16x16x32_bf16(
                           false, A, false, B0, (short)0, acc0, false, false);
                acc1 = __builtin_amdgcn_wmma_f32_16x16x32_bf16(
                           false, A, false, B1, (short)0, acc1, false, false);
            }
            const float bias0 = b2[(size_t)c * D_ + n0 + col];
            const float bias1 = b2[(size_t)c * D_ + n0 + 16 + col];
            const int d0 = n0 + col, d1 = n0 + 16 + col;
            const float w300 = __builtin_nontemporal_load(W3 + (size_t)c * D_ * 2 + d0 * 2 + 0);
            const float w301 = __builtin_nontemporal_load(W3 + (size_t)c * D_ * 2 + d0 * 2 + 1);
            const float w310 = __builtin_nontemporal_load(W3 + (size_t)c * D_ * 2 + d1 * 2 + 0);
            const float w311 = __builtin_nontemporal_load(W3 + (size_t)c * D_ * 2 + d1 * 2 + 1);
            #pragma unroll
            for (int v = 0; v < 8; ++v) {
                float h0 = acc0[v] + bias0; h0 = h0 > 0.f ? h0 : 0.f;
                float h1v = acc1[v] + bias1; h1v = h1v > 0.f ? h1v : 0.f;
                o0[v] += h0 * w300 + h1v * w310;     // layer 3 folded in (E=2)
                o1[v] += h0 * w301 + h1v * w311;
            }
        }
    }

    // ------- Reduce layer-3 partials across the 16 d-lanes, write out ------
    #pragma unroll
    for (int v = 0; v < 8; ++v) {
        #pragma unroll
        for (int m = 1; m < 16; m <<= 1) {
            o0[v] += __shfl_xor(o0[v], m, 16);
            o1[v] += __shfl_xor(o1[v], m, 16);
        }
    }
    if (col == 0) {
        const float bb0 = b3[c * 2 + 0];
        const float bb1 = b3[c * 2 + 1];
        #pragma unroll
        for (int v = 0; v < 8; ++v) {
            const int m = mbase + v + hi * 8;
            float2 rr; rr.x = o0[v] + bb0; rr.y = o1[v] + bb1;
            *(float2*)(out + (size_t)m * C_ * 2 + c * 2) = rr;   // out[b][c][:]
        }
    }
}

extern "C" void kernel_launch(void* const* d_in, const int* in_sizes, int n_in,
                              void* d_out, int out_size, void* d_ws, size_t ws_size,
                              hipStream_t stream) {
    (void)in_sizes; (void)n_in; (void)out_size; (void)d_ws; (void)ws_size;
    const float* x  = (const float*)d_in[0];
    const float* W1 = (const float*)d_in[1];
    const float* b1 = (const float*)d_in[2];
    const float* W2 = (const float*)d_in[3];
    const float* b2 = (const float*)d_in[4];
    const float* W3 = (const float*)d_in[5];
    const float* b3 = (const float*)d_in[6];
    float* out = (float*)d_out;

    // Allow >64 KB dynamic LDS; idempotent and capture-safe.
    (void)hipFuncSetAttribute((const void*)TMM_79413945303453_kernel,
                              hipFuncAttributeMaxDynamicSharedMemorySize,
                              (int)LDS_BYTES);

    TMM_79413945303453_kernel<<<C_, THREADS, LDS_BYTES, stream>>>(
        x, W1, b1, W2, b2, W3, b3, out);
}
